// MyViT_36223754175130
// MI455X (gfx1250) — matrix operand order repaired and checked by reading, not compiled, not run on hardware
//
#include <hip/hip_runtime.h>
#include <hip/hip_bf16.h>
#include <math.h>

// ---------------- dims (ViT-Base, B=32) ----------------
#define B_    32
#define S_    197
#define SPAD_ 224      // scores row padded to multiple of 32 (zero-filled pad)
#define D_    768
#define NH_   12
#define DH_   64
#define L_    12
#define MLP_  3072
#define OUT_  1000
#define NPX_  196      // patches per image (14*14)
#define PSQ_  768      // C*PS*PS = 3*16*16

typedef __attribute__((ext_vector_type(16))) __bf16 v16bf;
typedef __attribute__((ext_vector_type(8)))  __bf16 v8bf;
typedef __attribute__((ext_vector_type(8)))  float  v8f;

__device__ __forceinline__ unsigned int f32_to_bf16_rne(float x) {
  unsigned int u = __float_as_uint(x);
  return (u + 0x7FFFu + ((u >> 16) & 1u)) >> 16;
}
__device__ __forceinline__ unsigned int pack_bf16x2(float a, float b) {
  return f32_to_bf16_rne(a) | (f32_to_bf16_rne(b) << 16);
}

// ============================================================
// Generic batched GEMM: C[b] = alpha * A[b] (MxK) * B[b] (KxN) + bias (+epi)
// bf16 WMMA, f32 accumulate. REQUIRES K % 32 == 0 (all call sites satisfy).
// Block = 256 threads = 8 wave32. Block tile 32(M) x 64(N); wave (wm,wn)
// owns a 16x16 tile. LDS fragments are double-buffered and PRE-SWIZZLED
// into the WMMA VGPR layout, so lanes read 32B fragments as 2x ds_load_b128.
// All LDS accesses index the __shared__ arrays DIRECTLY (no pointer arrays)
// so address-space inference keeps them as ds_* ops, not flat_*.
//   A frag: lane = (m&15) + (k&8 ? 16:0), elem = ((k&16)>>1)|(k&7)
//   B frag: lane = (n&15) + (k&8 ? 16:0), elem = ((k&16)>>1)|(k&7)
// M/N edges: clamped-pointer loads + branchless 0/1 mask multiply.
// epi: 0 = store, 1 = store w/ exact GELU, 2 = C += result (residual add)
// ============================================================
template<bool TRANSB>
__global__ __launch_bounds__(256)
void gemm_bf16_wmma(const float* __restrict__ A, const float* __restrict__ Bm,
                    const float* __restrict__ bias, float* __restrict__ C,
                    int M, int N, int K, int lda, int ldb, int ldc,
                    long long sAo, long long sAi, long long sBo, long long sBi,
                    long long sCo, long long sCi, long long sBias,
                    int inner, float alpha, int epi)
{
  // [buf][tile][lane][elem], elem dim padded 16->24 (48B row, 16B aligned)
  __shared__ __attribute__((aligned(16))) unsigned short Afrag[2][2][32][24];
  __shared__ __attribute__((aligned(16))) unsigned short Bfrag[2][4][32][24];

  int bz = blockIdx.z;
  int bo = bz / inner, bi = bz - bo * inner;
  A  += bo * sAo + bi * sAi;
  Bm += bo * sBo + bi * sBi;
  C  += bo * sCo + bi * sCi;
  const float* biasp = bias ? (bias + bi * sBias) : (const float*)0;

  int tid  = threadIdx.x;
  int lane = tid & 31;
  int wid  = tid >> 5;
  int wm   = wid >> 2;       // 0..1
  int wn   = wid & 3;        // 0..3
  int tm0  = blockIdx.y * 32;
  int tn0  = blockIdx.x * 64;

  // ---- A staging coords: thread t loads A(ar, ac..ac+3) of the 32x32 tile ----
  int ar = tid >> 3;                  // 0..31
  int ac = (tid & 7) << 2;            // 0,4,...,28
  int a_tile = ar >> 4;
  int a_lane = (ar & 15) + ((ac & 8) ? 16 : 0);
  int a_e0   = ((ac & 16) >> 1) | (ac & 7);

  // ---- B staging coords: thread t loads B(bk0..bk0+7, bn) of the 32x64 tile ----
  int bn  = tid & 63;                 // 0..63
  int bk0 = (tid >> 6) << 3;          // 0,8,16,24
  int b_wn   = bn >> 4;
  int b_lane = (bn & 15) + ((bk0 & 8) ? 16 : 0);
  int b_e0   = (bk0 & 16) >> 1;       // 0 or 8 -> 16B aligned

  // ---- running global pointers + branchless edge masks ----
  int gm  = tm0 + ar;
  int gmc = (gm < M) ? gm : (M - 1);
  float amask = (gm < M) ? 1.0f : 0.0f;
  const float* pA = A + (long long)gmc * lda + ac;

  int gn  = tn0 + bn;
  int gnc = (gn < N) ? gn : (N - 1);
  float bmask = (gn < N) ? 1.0f : 0.0f;
  const float* pB = TRANSB ? (Bm + (long long)gnc * ldb + bk0)
                           : (Bm + (long long)bk0 * ldb + gnc);
  const long long pBstep = TRANSB ? 32 : (long long)32 * ldb;

  auto stage = [&](int buf) {
    float v0 = pA[0], v1 = pA[1], v2 = pA[2], v3 = pA[3];
    pA += 32;
    uint2 wa;
    wa.x = pack_bf16x2(v0 * amask, v1 * amask);
    wa.y = pack_bf16x2(v2 * amask, v3 * amask);
    *(uint2*)&Afrag[buf][a_tile][a_lane][a_e0] = wa;       // ds_store_b64

    float u[8];
    if (TRANSB) {
      #pragma unroll
      for (int i = 0; i < 8; ++i) u[i] = pB[i];
    } else {
      #pragma unroll
      for (int i = 0; i < 8; ++i) u[i] = pB[(long long)i * ldb];
    }
    pB += pBstep;
    uint4 wb;
    wb.x = pack_bf16x2(u[0] * bmask, u[1] * bmask);
    wb.y = pack_bf16x2(u[2] * bmask, u[3] * bmask);
    wb.z = pack_bf16x2(u[4] * bmask, u[5] * bmask);
    wb.w = pack_bf16x2(u[6] * bmask, u[7] * bmask);
    *(uint4*)&Bfrag[buf][b_wn][b_lane][b_e0] = wb;         // ds_store_b128
  };

  v8f acc = {};
  const int T = K >> 5;                      // K % 32 == 0 by construction

  stage(0);
  for (int i = 0; i < T; ++i) {
    __syncthreads();                         // buf (i&1) staged; prior reads done
    if (i + 1 < T) stage((i + 1) & 1);       // prefetch next tile into other buf
    int p = i & 1;
    union { v16bf v; v8bf h[2]; } af, bf;
    af.h[0] = *(const v8bf*)&Afrag[p][wm][lane][0];        // ds_load_b128
    af.h[1] = *(const v8bf*)&Afrag[p][wm][lane][8];        // ds_load_b128
    bf.h[0] = *(const v8bf*)&Bfrag[p][wn][lane][0];        // ds_load_b128
    bf.h[1] = *(const v8bf*)&Bfrag[p][wn][lane][8];        // ds_load_b128
    acc = __builtin_amdgcn_wmma_f32_16x16x32_bf16(false, af.v, false, bf.v,
                                                  (short)0, acc, false, false);
  }

  // ---- epilogue: D VGPR e holds row (e + hi*8), col = lane&15 ----
  int cn = tn0 + 16 * wn + (lane & 15);
  int mb = tm0 + 16 * wm + ((lane >> 4) ? 8 : 0);
  if (cn < N) {
    float bv = biasp ? biasp[cn] : 0.0f;
    #pragma unroll
    for (int e = 0; e < 8; ++e) {
      int cm = mb + e;
      if (cm < M) {
        float v = acc[e] * alpha + bv;
        if (epi == 1) v = 0.5f * v * (1.0f + erff(v * 0.70710678118f)); // exact GELU
        long long off = (long long)cm * ldc + cn;
        if (epi == 2) C[off] += v; else C[off] = v;
      }
    }
  }
}

// ---------------- patchify: x[B,3,224,224] -> P[B*196, 768] ----------------
__global__ __launch_bounds__(256)
void patchify_k(const float* __restrict__ x, float* __restrict__ P)
{
  long long idx = (long long)blockIdx.x * 256 + threadIdx.x;
  const long long total = (long long)B_ * NPX_ * PSQ_;
  if (idx >= total) return;
  int col = (int)(idx % PSQ_);
  long long row = idx / PSQ_;
  int n  = (int)(row / NPX_);
  int pij = (int)(row % NPX_);
  int pi = pij / 14, pj = pij % 14;
  int c  = col >> 8;
  int r  = col & 255;
  int ph = r >> 4, pw = r & 15;
  P[idx] = x[(((long long)n * 3 + c) * 224 + (pi * 16 + ph)) * 224 + (pj * 16 + pw)];
}

// ---- assemble tokens: cls + patch-embeds, plus sinusoidal positional emb ----
__global__ __launch_bounds__(256)
void assemble_k(const float* __restrict__ tmp, const float* __restrict__ cls,
                float* __restrict__ tokens)
{
  long long idx = (long long)blockIdx.x * 256 + threadIdx.x;
  const long long total = (long long)B_ * S_ * D_;
  if (idx >= total) return;
  int d = (int)(idx % D_);
  long long ns = idx / D_;
  int s = (int)(ns % S_);
  long long n = ns / S_;
  float jf = (d & 1) ? (float)(d - 1) : (float)d;
  float angle = (float)s / powf(10000.0f, jf / (float)D_);
  float pos = (d & 1) ? cosf(angle) : sinf(angle);
  float v = (s == 0) ? cls[d] : tmp[((long long)n * NPX_ + (s - 1)) * D_ + d];
  tokens[idx] = v + pos;
}

// ---------------- layernorm over last dim (=768), one block per row ----------------
__global__ __launch_bounds__(256)
void layernorm_k(const float* __restrict__ x, const float* __restrict__ g,
                 const float* __restrict__ b, float* __restrict__ y)
{
  __shared__ float rs[256], rs2[256];
  long long row = blockIdx.x;
  const float* xr = x + row * D_;
  float* yr = y + row * D_;
  int tid = threadIdx.x;
  float s = 0.0f, s2 = 0.0f;
  for (int i = tid; i < D_; i += 256) { float v = xr[i]; s += v; s2 += v * v; }
  rs[tid] = s; rs2[tid] = s2; __syncthreads();
  for (int st = 128; st > 0; st >>= 1) {
    if (tid < st) { rs[tid] += rs[tid + st]; rs2[tid] += rs2[tid + st]; }
    __syncthreads();
  }
  float mu  = rs[0] * (1.0f / D_);
  float var = rs2[0] * (1.0f / D_) - mu * mu;
  float inv = rsqrtf(var + 1e-5f);
  for (int i = tid; i < D_; i += 256) yr[i] = (xr[i] - mu) * inv * g[i] + b[i];
}

// ---- row softmax over [0,R), rows have leading dim Rstride; pad cols zeroed ----
__global__ __launch_bounds__(256)
void softmax_k(const float* __restrict__ in, float* __restrict__ out,
               int R, int Rstride)
{
  __shared__ float red[256];
  long long row = blockIdx.x;
  const float* xr = in + row * (long long)Rstride;
  float* yr = out + row * (long long)Rstride;
  int tid = threadIdx.x;
  float m = -3.402823466e38f;
  for (int i = tid; i < R; i += 256) m = fmaxf(m, xr[i]);
  red[tid] = m; __syncthreads();
  for (int st = 128; st > 0; st >>= 1) {
    if (tid < st) red[tid] = fmaxf(red[tid], red[tid + st]);
    __syncthreads();
  }
  m = red[0]; __syncthreads();
  float s = 0.0f;
  for (int i = tid; i < R; i += 256) s += expf(xr[i] - m);
  red[tid] = s; __syncthreads();
  for (int st = 128; st > 0; st >>= 1) {
    if (tid < st) red[tid] += red[tid + st];
    __syncthreads();
  }
  float inv = 1.0f / red[0];
  for (int i = tid; i < R; i += 256) yr[i] = expf(xr[i] - m) * inv;
  for (int i = R + tid; i < Rstride; i += 256) yr[i] = 0.0f;  // zero the K-pad
}

// ---------------- host-side helpers ----------------
static inline void launch_gemm(hipStream_t stream,
                               const float* A, const float* Bm, const float* bias, float* C,
                               int M, int N, int K, int lda, int ldb, int ldc,
                               long long sAo, long long sAi, long long sBo, long long sBi,
                               long long sCo, long long sCi, long long sBias,
                               int outer, int inner, int transB, float alpha, int epi)
{
  dim3 grid((N + 63) / 64, (M + 31) / 32, outer * inner);
  if (transB)
    gemm_bf16_wmma<true><<<grid, 256, 0, stream>>>(A, Bm, bias, C, M, N, K, lda, ldb, ldc,
                                                   sAo, sAi, sBo, sBi, sCo, sCi, sBias,
                                                   inner, alpha, epi);
  else
    gemm_bf16_wmma<false><<<grid, 256, 0, stream>>>(A, Bm, bias, C, M, N, K, lda, ldb, ldc,
                                                    sAo, sAi, sBo, sBi, sCo, sCi, sBias,
                                                    inner, alpha, epi);
}

extern "C" void kernel_launch(void* const* d_in, const int* in_sizes, int n_in,
                              void* d_out, int out_size, void* d_ws, size_t ws_size,
                              hipStream_t stream)
{
  (void)in_sizes; (void)n_in; (void)out_size; (void)ws_size;
  const float* x       = (const float*)d_in[0];
  const float* w_embed = (const float*)d_in[1];
  const float* b_embed = (const float*)d_in[2];
  const float* cls_tok = (const float*)d_in[3];
  const float* ln1_g   = (const float*)d_in[4];
  const float* ln1_b   = (const float*)d_in[5];
  const float* wq      = (const float*)d_in[6];
  const float* bq      = (const float*)d_in[7];
  const float* wk      = (const float*)d_in[8];
  const float* bk      = (const float*)d_in[9];
  const float* wv      = (const float*)d_in[10];
  const float* bv      = (const float*)d_in[11];
  const float* ln2_g   = (const float*)d_in[12];
  const float* ln2_b   = (const float*)d_in[13];
  const float* w1      = (const float*)d_in[14];
  const float* b1      = (const float*)d_in[15];
  const float* w2      = (const float*)d_in[16];
  const float* b2      = (const float*)d_in[17];
  const float* w_head  = (const float*)d_in[18];
  const float* b_head  = (const float*)d_in[19];
  float* out = (float*)d_out;

  // ---- workspace layout (floats), with temporal reuse of the big pool ----
  float* ws = (float*)d_ws;
  const long long TOK = (long long)B_ * S_ * D_;          // 4,841,472
  const long long QKV = (long long)B_ * NH_ * S_ * DH_;   // 4,841,472
  float* tokens  = ws;                 // persistent residual stream
  float* hbuf    = ws + TOK;           // LN output / logits reuse
  float* pool    = ws + 2 * TOK;       // time-shared pool
  float* q       = pool;
  float* kbuf    = pool + QKV;
  float* vbuf    = pool + 2 * QKV;
  float* scores  = pool + 3 * QKV;     // [B, NH, S, SPAD]
  float* m1      = pool;               // MLP hidden (disjoint in time)
  float* patches = pool;               // only used at start
  float* tmp     = pool + (long long)B_ * NPX_ * D_;
  float* logits  = hbuf;

  const long long sTokN  = (long long)S_ * D_;      // per-image token stride
  const long long sHeadQ = (long long)S_ * DH_;     // per-head stride in q/k/v
  const long long sHeadS = (long long)S_ * SPAD_;   // per-head stride in scores

  // 1) patchify + patch embedding
  {
    long long total = (long long)B_ * NPX_ * PSQ_;
    patchify_k<<<dim3((unsigned)((total + 255) / 256)), 256, 0, stream>>>(x, patches);
  }
  launch_gemm(stream, patches, w_embed, b_embed, tmp,
              B_ * NPX_, D_, PSQ_, PSQ_, D_, D_,
              0, 0, 0, 0, 0, 0, 0, 1, 1, 0, 1.0f, 0);
  {
    long long total = TOK;
    assemble_k<<<dim3((unsigned)((total + 255) / 256)), 256, 0, stream>>>(tmp, cls_tok, tokens);
  }

  // 2) transformer layers
  for (int l = 0; l < L_; ++l) {
    const float* wql = wq + (long long)l * NH_ * DH_ * DH_;
    const float* wkl = wk + (long long)l * NH_ * DH_ * DH_;
    const float* wvl = wv + (long long)l * NH_ * DH_ * DH_;
    const float* bql = bq + (long long)l * NH_ * DH_;
    const float* bkl = bk + (long long)l * NH_ * DH_;
    const float* bvl = bv + (long long)l * NH_ * DH_;

    // LN1: tokens -> hbuf
    layernorm_k<<<dim3(B_ * S_), 256, 0, stream>>>(tokens, ln1_g + l * D_, ln1_b + l * D_, hbuf);

    // Q/K/V: per (n, head) GEMM  [S x 64] = [S x 64] @ [64 x 64]
    launch_gemm(stream, hbuf, wql, bql, q,
                S_, DH_, DH_, D_, DH_, DH_,
                sTokN, DH_, 0, (long long)DH_ * DH_,
                (long long)NH_ * sHeadQ, sHeadQ, DH_,
                B_, NH_, 0, 1.0f, 0);
    launch_gemm(stream, hbuf, wkl, bkl, kbuf,
                S_, DH_, DH_, D_, DH_, DH_,
                sTokN, DH_, 0, (long long)DH_ * DH_,
                (long long)NH_ * sHeadQ, sHeadQ, DH_,
                B_, NH_, 0, 1.0f, 0);
    launch_gemm(stream, hbuf, wvl, bvl, vbuf,
                S_, DH_, DH_, D_, DH_, DH_,
                sTokN, DH_, 0, (long long)DH_ * DH_,
                (long long)NH_ * sHeadQ, sHeadQ, DH_,
                B_, NH_, 0, 1.0f, 0);

    // scores = (q @ k^T) / 8 : per (n,head) [S x S], transB, ldc = SPAD
    launch_gemm(stream, q, kbuf, (const float*)0, scores,
                S_, S_, DH_, DH_, DH_, SPAD_,
                (long long)NH_ * sHeadQ, sHeadQ,
                (long long)NH_ * sHeadQ, sHeadQ,
                (long long)NH_ * sHeadS, sHeadS, 0,
                B_, NH_, 1, 0.125f, 0);

    // softmax over t (also zero-fills pad columns [S, SPAD))
    softmax_k<<<dim3(B_ * NH_ * S_), 256, 0, stream>>>(scores, scores, S_, SPAD_);

    // tokens += attn @ v : per (n,head) [S x 64], K padded to SPAD (zeros)
    launch_gemm(stream, scores, vbuf, (const float*)0, tokens,
                S_, DH_, SPAD_, SPAD_, DH_, D_,
                (long long)NH_ * sHeadS, sHeadS,
                (long long)NH_ * sHeadQ, sHeadQ,
                sTokN, DH_, 0,
                B_, NH_, 0, 1.0f, 2);

    // LN2: tokens -> hbuf
    layernorm_k<<<dim3(B_ * S_), 256, 0, stream>>>(tokens, ln2_g + l * D_, ln2_b + l * D_, hbuf);

    // MLP1: m1 = gelu(hbuf @ w1 + b1)   [B*S x 3072]
    launch_gemm(stream, hbuf, w1 + (long long)l * D_ * MLP_, b1 + (long long)l * MLP_, m1,
                B_ * S_, MLP_, D_, D_, MLP_, MLP_,
                0, 0, 0, 0, 0, 0, 0, 1, 1, 0, 1.0f, 1);

    // MLP2: tokens += m1 @ w2 + b2
    launch_gemm(stream, m1, w2 + (long long)l * MLP_ * D_, b2 + (long long)l * D_, tokens,
                B_ * S_, D_, MLP_, MLP_, D_, D_,
                0, 0, 0, 0, 0, 0, 0, 1, 1, 0, 1.0f, 2);
  }

  // 3) classification head on cls token rows (row n at tokens + n*S*D -> lda = S*D)
  launch_gemm(stream, tokens, w_head, b_head, logits,
              B_, OUT_, D_, (int)sTokN, OUT_, OUT_,
              0, 0, 0, 0, 0, 0, 0, 1, 1, 0, 1.0f, 0);

  // 4) final softmax -> d_out
  softmax_k<<<dim3(B_), 256, 0, stream>>>(logits, out, OUT_, OUT_);
}
